// CrossAttention_49246095015976
// MI455X (gfx1250) — matrix operand order repaired and checked by reading, not compiled, not run on hardware
//
#include <hip/hip_runtime.h>
#include <hip/hip_bf16.h>

typedef __attribute__((ext_vector_type(16))) _Float16 v16h;
typedef __attribute__((ext_vector_type(8)))  _Float16 v8h;
typedef __attribute__((ext_vector_type(8)))  float    v8f;

#define DEVINL static __device__ __forceinline__

// ---------------------------------------------------------------------------
// WMMA fragment helpers (wave32, V_WMMA_F32_16X16X32_F16)
// A-matrix 16x32 (MxK): lane<16 row M=lane, K = {0..7}(v0-3) + {16..23}(v4-7);
//                       lane>=16 same row, K = {8..15} + {24..31}.
// B-matrix 32x16 (KxN): lane holds col N=lane&15; same K split as A:
//                       lanes 0-15: K={0..7}+{16..23}, lanes 16-31: K={8..15}+{24..31}.
// C/D 16x16 f32: VGPR j -> M = j + (lane>=16 ? 8 : 0), N = lane&15.
// ---------------------------------------------------------------------------
DEVINL v16h load_a_frag(const _Float16* row_k0, int lane) {
  const _Float16* p = row_k0 + ((lane & 16) ? 8 : 0);
  v16h r;
  *reinterpret_cast<v8h*>(&r)       = *reinterpret_cast<const v8h*>(p);
  *(reinterpret_cast<v8h*>(&r) + 1) = *reinterpret_cast<const v8h*>(p + 16);
  return r;
}

DEVINL v16h load_b_frag(const _Float16* col_k0, int lane) {
  const _Float16* p = col_k0 + ((lane & 16) ? 8 : 0);
  v16h r;
  *reinterpret_cast<v8h*>(&r)       = *reinterpret_cast<const v8h*>(p);
  *(reinterpret_cast<v8h*>(&r) + 1) = *reinterpret_cast<const v8h*>(p + 16);
  return r;
}

DEVINL v8f wmma_f16(v16h a, v16h b, v8f c) {
  return __builtin_amdgcn_wmma_f32_16x16x32_f16(false, a, false, b, (short)0, c,
                                                false, false);
}

// ---------------------------------------------------------------------------
// Width-16 XOR butterfly reductions via v_permlane16_b32 (pure VALU, no LDS).
// ---------------------------------------------------------------------------
#if __has_builtin(__builtin_amdgcn_permlane16)
#define CA_HAVE_PERMLANE16 1
DEVINL float ca_pl16(float v, unsigned s0, unsigned s1) {
  union { float f; unsigned u; } a, r;
  a.f = v;
  r.u = __builtin_amdgcn_permlane16(a.u, a.u, s0, s1, false, false);
  return r.f;
}
#endif

DEVINL float red_max16(float v) {
#ifdef CA_HAVE_PERMLANE16
  v = fmaxf(v, ca_pl16(v, 0x67452301u, 0xEFCDAB89u));  // xor 1
  v = fmaxf(v, ca_pl16(v, 0x54761032u, 0xDCFE98BAu));  // xor 2
  v = fmaxf(v, ca_pl16(v, 0x32107654u, 0xBA98FEDCu));  // xor 4
  v = fmaxf(v, ca_pl16(v, 0xFEDCBA98u, 0x76543210u));  // xor 8
#else
#pragma unroll
  for (int off = 1; off < 16; off <<= 1) v = fmaxf(v, __shfl_xor(v, off, 16));
#endif
  return v;
}

DEVINL float red_sum16(float v) {
#ifdef CA_HAVE_PERMLANE16
  v += ca_pl16(v, 0x67452301u, 0xEFCDAB89u);
  v += ca_pl16(v, 0x54761032u, 0xDCFE98BAu);
  v += ca_pl16(v, 0x32107654u, 0xBA98FEDCu);
  v += ca_pl16(v, 0xFEDCBA98u, 0x76543210u);
#else
#pragma unroll
  for (int off = 1; off < 16; off <<= 1) v += __shfl_xor(v, off, 16);
#endif
  return v;
}

// ---------------------------------------------------------------------------
// Stage 0: f32 -> f16 conversion / transpose of weights
// ---------------------------------------------------------------------------
__global__ void ca_f32_to_f16_v8(const float* __restrict__ in,
                                 _Float16* __restrict__ out, long long n8) {
  long long i = (long long)blockIdx.x * blockDim.x + threadIdx.x;
  if (i >= n8) return;
  const float4* p = reinterpret_cast<const float4*>(in + i * 8);
  float4 f0 = p[0];
  float4 f1 = p[1];
  v8h o;
  o[0] = (_Float16)f0.x; o[1] = (_Float16)f0.y;
  o[2] = (_Float16)f0.z; o[3] = (_Float16)f0.w;
  o[4] = (_Float16)f1.x; o[5] = (_Float16)f1.y;
  o[6] = (_Float16)f1.z; o[7] = (_Float16)f1.w;
  *reinterpret_cast<v8h*>(out + i * 8) = o;
}

// in [K,N] f32 row-major -> out [N,K] f16 row-major
__global__ void ca_transpose_f32_f16(const float* __restrict__ in,
                                     _Float16* __restrict__ out, int K, int N) {
  long long idx = (long long)blockIdx.x * blockDim.x + threadIdx.x;
  if (idx >= (long long)K * N) return;
  int k = (int)(idx % K);
  int n = (int)(idx / K);
  out[(size_t)n * K + k] = (_Float16)in[(size_t)k * N + n];
}

// ---------------------------------------------------------------------------
// Generic GEMM: C[M,N] = A[M,K] (f16, row-major) x Bt[N,K] (f16, B transposed)
// Each wave computes a 32(M) x 64(N) tile; K in chunks of 32; B fragments are
// shared by two A fragments (8 WMMAs per 6 fragment loads).
// MODE 0: store f16 row-major [M,N]
// MODE 1: store f16 "per-head transposed": row (b*1024 + col), col = row%1024
// MODE 2: store f32 [M,N] with bias
// ---------------------------------------------------------------------------
template <int MODE>
__global__ __launch_bounds__(128) void ca_gemm32x64(
    const _Float16* __restrict__ A, const _Float16* __restrict__ Bt,
    void* __restrict__ Cout, const float* __restrict__ bias, int M, int N,
    int K) {
  const int lane = threadIdx.x & 31;
  const int wave = (blockIdx.x * blockDim.x + threadIdx.x) >> 5;
  const int ntile64 = N >> 6;
  const int tm = wave / ntile64;
  const int tn = wave % ntile64;
  if (tm * 32 >= M) return;
  const int l15 = lane & 15;
  const int mh = (lane & 16) ? 8 : 0;

  const _Float16* arow0 = A + (size_t)(tm * 32 + l15) * K;
  const _Float16* arow1 = arow0 + (size_t)16 * K;
  const _Float16* brow[4];
#pragma unroll
  for (int t = 0; t < 4; ++t)
    brow[t] = Bt + (size_t)(tn * 64 + t * 16 + l15) * K;

  v8f acc[2][4] = {};
  for (int kc = 0; kc < K; kc += 32) {
    v16h a0 = load_a_frag(arow0 + kc, lane);
    v16h a1 = load_a_frag(arow1 + kc, lane);
#pragma unroll
    for (int t = 0; t < 4; ++t) {
      v16h b = load_b_frag(brow[t] + kc, lane);
      acc[0][t] = wmma_f16(a0, b, acc[0][t]);
      acc[1][t] = wmma_f16(a1, b, acc[1][t]);
    }
  }

#pragma unroll
  for (int r = 0; r < 2; ++r) {
    const int mbase = tm * 32 + r * 16 + mh;
    if (MODE == 0) {
      _Float16* out = (_Float16*)Cout;
#pragma unroll
      for (int t = 0; t < 4; ++t) {
        int n = tn * 64 + t * 16 + l15;
#pragma unroll
        for (int j = 0; j < 8; ++j)
          out[(size_t)(mbase + j) * N + n] = (_Float16)acc[r][t][j];
      }
    } else if (MODE == 1) {
      _Float16* out = (_Float16*)Cout;
      int bb = mbase >> 10;   // batch index (1024 ctx rows per batch)
      int n0 = mbase & 1023;  // ctx row within batch
#pragma unroll
      for (int t = 0; t < 4; ++t) {
        int jcol = tn * 64 + t * 16 + l15;  // = h*64 + d
        v8h pk;
#pragma unroll
        for (int j = 0; j < 8; ++j) pk[j] = (_Float16)acc[r][t][j];
        *reinterpret_cast<v8h*>(out + ((size_t)(bb * 1024 + jcol)) * 1024 +
                                n0) = pk;
      }
    } else {
      float* out = (float*)Cout;
#pragma unroll
      for (int t = 0; t < 4; ++t) {
        int n = tn * 64 + t * 16 + l15;
        float bv = bias ? bias[n] : 0.0f;
#pragma unroll
        for (int j = 0; j < 8; ++j)
          out[(size_t)(mbase + j) * N + n] = acc[r][t][j] + bv;
      }
    }
  }
}

// ---------------------------------------------------------------------------
// Flash attention: one wave handles 32 queries x one head (dh=64), streaming
// keys in chunks of 64. Two 16-query row groups share every K/V fragment
// (32 FLOP/B from the memory path; 32 WMMAs per 16KB of K/V).
// Q  [B*2048, 1024] f16 row-major (head h at col h*64)
// Kh [B*1024, 1024] f16 row-major
// Vt [B*16*64, 1024] f16 : row (b*16+h)*64+d, col = key index n
// O  [B*2048, 1024] f16 row-major
// ---------------------------------------------------------------------------
__global__ __launch_bounds__(128) void ca_attention(
    const _Float16* __restrict__ Q, const _Float16* __restrict__ Kh,
    const _Float16* __restrict__ Vt, _Float16* __restrict__ O) {
  __shared__ __align__(16) _Float16 Plds[4][32 * 64];  // 16 KB / block
  const int lane = threadIdx.x & 31;
  const int wl = threadIdx.x >> 5;
  const int wave = blockIdx.x * 4 + wl;
  const int qt = wave & 63;  // 64 q-tiles of 32 per (b,h)
  const int bh = wave >> 6;  // 0..63
  const int h = bh & 15;
  const int b = bh >> 4;
  const int l15 = lane & 15;
  const int mh = (lane & 16) ? 8 : 0;
  const float scale = 0.125f;  // 1/sqrt(64)

  // preload Q A-fragments: 2 row groups x (d=0..31, d=32..63)
  const _Float16* qrow0 = Q + (size_t)(b * 2048 + qt * 32 + l15) * 1024 + h * 64;
  const _Float16* qrow1 = qrow0 + (size_t)16 * 1024;
  v16h qa[2][2];
  qa[0][0] = load_a_frag(qrow0, lane);
  qa[0][1] = load_a_frag(qrow0 + 32, lane);
  qa[1][0] = load_a_frag(qrow1, lane);
  qa[1][1] = load_a_frag(qrow1 + 32, lane);

  v8f Oacc[2][4] = {};
  float runmax[2][8], runsum[2][8];
#pragma unroll
  for (int r = 0; r < 2; ++r)
#pragma unroll
    for (int j = 0; j < 8; ++j) {
      runmax[r][j] = -__builtin_inff();
      runsum[r][j] = 0.0f;
    }

  const _Float16* kbase = Kh + (size_t)(b * 1024) * 1024 + h * 64;
  const int vbaserow = (b * 16 + h) * 64;
  _Float16* pl = Plds[wl];

  for (int nc = 0; nc < 1024; nc += 64) {
    // ---- scores: S[32q x 64k] = Q (32x64) x K^T; K frags shared by groups --
    v8f s[2][4] = {};
#pragma unroll
    for (int t = 0; t < 4; ++t) {
      const _Float16* kr = kbase + (size_t)(nc + t * 16 + l15) * 1024;
      v16h kb0 = load_b_frag(kr, lane);
      v16h kb1 = load_b_frag(kr + 32, lane);
      s[0][t] = wmma_f16(qa[0][0], kb0, s[0][t]);
      s[1][t] = wmma_f16(qa[1][0], kb0, s[1][t]);
      s[0][t] = wmma_f16(qa[0][1], kb1, s[0][t]);
      s[1][t] = wmma_f16(qa[1][1], kb1, s[1][t]);
    }
    // ---- online softmax per row group (C rows live in one 16-lane half) ---
#pragma unroll
    for (int r = 0; r < 2; ++r) {
      float corr[8];
#pragma unroll
      for (int j = 0; j < 8; ++j) {
        float a0 = s[r][0][j] * scale;
        float a1 = s[r][1][j] * scale;
        float a2 = s[r][2][j] * scale;
        float a3 = s[r][3][j] * scale;
        float mx = red_max16(fmaxf(fmaxf(a0, a1), fmaxf(a2, a3)));
        float nm = fmaxf(runmax[r][j], mx);
        float c = __expf(runmax[r][j] - nm);
        runmax[r][j] = nm;
        float p0 = __expf(a0 - nm);
        float p1 = __expf(a1 - nm);
        float p2 = __expf(a2 - nm);
        float p3 = __expf(a3 - nm);
        float sm = red_sum16((p0 + p1) + (p2 + p3));
        runsum[r][j] = runsum[r][j] * c + sm;
        corr[j] = c;
        s[r][0][j] = p0;
        s[r][1][j] = p1;
        s[r][2][j] = p2;
        s[r][3][j] = p3;
      }
#pragma unroll
      for (int dt = 0; dt < 4; ++dt)
#pragma unroll
        for (int j = 0; j < 8; ++j) Oacc[r][dt][j] *= corr[j];
    }

    // ---- restage P (C layout -> A layout) through LDS ----
    __syncthreads();
#pragma unroll
    for (int r = 0; r < 2; ++r)
#pragma unroll
      for (int j = 0; j < 8; ++j)
#pragma unroll
        for (int t = 0; t < 4; ++t)
          pl[(r * 16 + mh + j) * 64 + t * 16 + l15] = (_Float16)s[r][t][j];
    __syncthreads();
    v16h pa[2][2];
#pragma unroll
    for (int r = 0; r < 2; ++r) {
      const _Float16* prow = pl + (r * 16 + l15) * 64;
      pa[r][0] = load_a_frag(prow, lane);
      pa[r][1] = load_a_frag(prow + 32, lane);
    }
    // ---- O += P (32x64) x V (64x64); V frags shared by groups ----
#pragma unroll
    for (int dt = 0; dt < 4; ++dt) {
      const _Float16* vrow =
          Vt + (size_t)(vbaserow + dt * 16 + l15) * 1024 + nc;
      v16h vb0 = load_b_frag(vrow, lane);
      v16h vb1 = load_b_frag(vrow + 32, lane);
      Oacc[0][dt] = wmma_f16(pa[0][0], vb0, Oacc[0][dt]);
      Oacc[1][dt] = wmma_f16(pa[1][0], vb0, Oacc[1][dt]);
      Oacc[0][dt] = wmma_f16(pa[0][1], vb1, Oacc[0][dt]);
      Oacc[1][dt] = wmma_f16(pa[1][1], vb1, Oacc[1][dt]);
    }
  }

  // epilogue: divide by row sum, store f16
#pragma unroll
  for (int r = 0; r < 2; ++r) {
    _Float16* orow =
        O + (size_t)(b * 2048 + qt * 32 + r * 16) * 1024 + h * 64;
#pragma unroll
    for (int j = 0; j < 8; ++j) {
      float inv = 1.0f / runsum[r][j];
#pragma unroll
      for (int dt = 0; dt < 4; ++dt)
        orow[(size_t)(mh + j) * 1024 + dt * 16 + l15] =
            (_Float16)(Oacc[r][dt][j] * inv);
    }
  }
}

// ---------------------------------------------------------------------------
extern "C" void kernel_launch(void* const* d_in, const int* in_sizes, int n_in,
                              void* d_out, int out_size, void* d_ws,
                              size_t ws_size, hipStream_t stream) {
  (void)in_sizes; (void)n_in; (void)out_size; (void)ws_size;
  const float* x   = (const float*)d_in[0];  // [4,2048,1024]
  const float* ctx = (const float*)d_in[1];  // [4,1024,768]
  const float* Wq  = (const float*)d_in[2];  // [1024,1024]
  const float* Wk  = (const float*)d_in[3];  // [768,1024]
  const float* Wv  = (const float*)d_in[4];  // [768,1024]
  const float* Wo  = (const float*)d_in[5];  // [1024,1024]
  const float* bo  = (const float*)d_in[6];  // [1024]

  char* w = (char*)d_ws;
  auto alloc = [&](size_t elems) {
    _Float16* p = (_Float16*)w;
    w += elems * sizeof(_Float16);
    return p;
  };
  _Float16* x_h   = alloc(8192ull * 1024);
  _Float16* ctx_h = alloc(4096ull * 768);
  _Float16* WqT   = alloc(1024ull * 1024);
  _Float16* WkT   = alloc(1024ull * 768);
  _Float16* WvT   = alloc(1024ull * 768);
  _Float16* WoT   = alloc(1024ull * 1024);
  _Float16* q_h   = alloc(8192ull * 1024);
  _Float16* k_h   = alloc(4096ull * 1024);
  _Float16* vT    = alloc(4096ull * 1024);
  _Float16* at_h  = alloc(8192ull * 1024);

  ca_f32_to_f16_v8<<<(1048576 + 255) / 256, 256, 0, stream>>>(x, x_h,
                                                              1048576LL);
  ca_f32_to_f16_v8<<<(393216 + 255) / 256, 256, 0, stream>>>(ctx, ctx_h,
                                                             393216LL);
  ca_transpose_f32_f16<<<(1048576 + 255) / 256, 256, 0, stream>>>(Wq, WqT,
                                                                  1024, 1024);
  ca_transpose_f32_f16<<<(786432 + 255) / 256, 256, 0, stream>>>(Wk, WkT, 768,
                                                                 1024);
  ca_transpose_f32_f16<<<(786432 + 255) / 256, 256, 0, stream>>>(Wv, WvT, 768,
                                                                 1024);
  ca_transpose_f32_f16<<<(1048576 + 255) / 256, 256, 0, stream>>>(Wo, WoT,
                                                                  1024, 1024);

  // q = x @ Wq   : [8192,1024] x [1024,1024]
  ca_gemm32x64<0><<<1024, 128, 0, stream>>>(x_h, WqT, q_h, nullptr, 8192, 1024,
                                            1024);
  // k = ctx @ Wk : [4096,768] x [768,1024]
  ca_gemm32x64<0><<<512, 128, 0, stream>>>(ctx_h, WkT, k_h, nullptr, 4096,
                                           1024, 768);
  // v = ctx @ Wv, stored transposed per head
  ca_gemm32x64<1><<<512, 128, 0, stream>>>(ctx_h, WvT, vT, nullptr, 4096, 1024,
                                           768);
  // fused softmax(q k^T / 8) v : 4096 waves of 32 queries each
  ca_attention<<<1024, 128, 0, stream>>>(q_h, k_h, vT, at_h);
  // out = attn @ Wo + bo (f32)
  ca_gemm32x64<2><<<1024, 128, 0, stream>>>(at_h, WoT, (float*)d_out, bo, 8192,
                                            1024, 1024);
}
